// SFIMoudle_11201274707991
// MI455X (gfx1250) — matrix operand order repaired and checked by reading, not compiled, not run on hardware
//
#include <hip/hip_runtime.h>
#include <hip/hip_bf16.h>
#include <stddef.h>

typedef __attribute__((ext_vector_type(16))) __bf16 v16bf;
typedef __attribute__((ext_vector_type(8)))  float  v8f;

#define Bn    2
#define Cc    64
#define Hh    128
#define Ww    128
#define HWp   16384       // Hh*Ww (power of 2: >>14, &16383)
#define WF    65          // rfft width
#define NFr   (Bn*Cc*Hh*WF)   // 1064960
#define NSp   (Bn*Cc*HWp)     // 2097152
#define Ltot  HWp
#define Mrows (Bn*Ltot)       // 32768
#define PI_F  3.14159265358979323846f

static __device__ __forceinline__ unsigned short f2bfu(float f) {
  union { float f; unsigned u; } c; c.f = f;
  return (unsigned short)((c.u + 0x7FFFu + ((c.u >> 16) & 1u)) >> 16);
}
static __device__ __forceinline__ unsigned packbf(float lo, float hi) {
  return (unsigned)f2bfu(lo) | ((unsigned)f2bfu(hi) << 16);
}
static __device__ __forceinline__ float siluf(float v) { return v / (1.f + __expf(-v)); }

// ---------------------------------------------------------------------------
// Generic GEMM: Out[M,N] = A[M,K] * W[N,K]^T (+bias), bf16 WMMA, f32 accum.
// Requirements: M % 16 == 0, K even. NCHW modes assume HW == HWp (16384).
// AMODE 0: A[m*lda + acol + k]   AMODE 1 (NCHW): A[(b*K+k)*HWp + p], m=b*HWp+p
// OMODE 0: Out[m*ldo + ocol + n] OMODE 1 (NCHW): Out[(b*N+n)*HWp + p]
// LDS tiles are stored pre-swizzled (per-lane-contiguous fragments):
//   A u32 index = j*256 + l*8 + r        -> lane l reads v16bf pA[j*32  + l]
//   B u32 index = j*1024 + (w*32+l)*8+q  -> lane l reads v16bf pB[j*128 + w*32 + l]
// ---------------------------------------------------------------------------
template <int AMODE, int OMODE>
__global__ __launch_bounds__(128) void k_gemm(
    const float* __restrict__ A, const float* __restrict__ Wt,
    const float* __restrict__ bias, float* __restrict__ Out,
    int M, int N, int K,
    int lda, int acol,
    int ldo, int ocol)
{
  __shared__ __align__(32) unsigned lA[512];    // 16 x 64 bf16
  __shared__ __align__(32) unsigned lB[2048];   // 64 x 64 bf16
  int tid  = threadIdx.x;
  int lane = tid & 31;
  int wv   = tid >> 5;
  int mtile = blockIdx.x * 16;
  int nbase = blockIdx.y * 64;
  v8f acc0 = {}, acc1 = {};
  const v16bf* pA = reinterpret_cast<const v16bf*>(lA);
  const v16bf* pB = reinterpret_cast<const v16bf*>(lB);
  for (int kk = 0; kk < K; kk += 64) {
    // stage A: 512 k-pairs, 4 per thread (clamped loads, branchless zeroing)
#pragma unroll
    for (int i = 0; i < 4; ++i) {
      int p = tid + i * 128;
      int j = p >> 8, rem = p & 255;
      int l = rem >> 3, r = rem & 7;
      int m = l & 15, g = l >> 4;
      int ka0 = ((r < 4) ? (2 * r) : (2 * r + 8)) + 8 * g;
      int k0 = kk + j * 32 + ka0;
      int gm = mtile + m;
      bool ok = k0 < K;
      int k0c = ok ? k0 : 0;
      float v0, v1;
      if (AMODE == 0) {
        int idx = gm * lda + acol + k0c;
        v0 = A[idx]; v1 = A[idx + 1];
      } else {
        int b = gm >> 14, pp = gm & (HWp - 1);
        int idx = (b * K + k0c) * HWp + pp;
        v0 = A[idx]; v1 = A[idx + HWp];
      }
      if (!ok) { v0 = 0.f; v1 = 0.f; }
      lA[j * 256 + l * 8 + r] = packbf(v0, v1);
    }
    // stage B: 2048 k-pairs, 16 per thread
#pragma unroll
    for (int i = 0; i < 16; ++i) {
      int p = tid + i * 128;
      int j = p >> 10, rem = p & 1023;
      int w = rem >> 8, l = (rem >> 3) & 31, q = rem & 7;
      int kb0 = ((q < 4) ? (2 * q) : (2 * q + 8)) + 8 * (l >> 4);
      int n = nbase + w * 16 + (l & 15);
      int k0 = kk + j * 32 + kb0;
      bool ok = (n < N) && (k0 < K);
      int nc = ok ? n : 0, kc = ok ? k0 : 0;
      int idx = nc * K + kc;
      float v0 = Wt[idx], v1 = Wt[idx + 1];
      if (!ok) { v0 = 0.f; v1 = 0.f; }
      lB[j * 1024 + (w * 32 + l) * 8 + q] = packbf(v0, v1);
    }
    if (AMODE == 0 && kk + 64 < K) {
      int gm = mtile + (tid & 15);
      __builtin_prefetch(&A[(size_t)gm * lda + acol + kk + 64], 0, 1);
    }
    __syncthreads();
    v16bf av, bv;
    av = pA[lane];            bv = pB[wv * 32 + lane];
    acc0 = __builtin_amdgcn_wmma_f32_16x16x32_bf16(false, av, false, bv,
                                                   (short)0, acc0, false, false);
    av = pA[32 + lane];       bv = pB[128 + wv * 32 + lane];
    acc1 = __builtin_amdgcn_wmma_f32_16x16x32_bf16(false, av, false, bv,
                                                   (short)0, acc1, false, false);
    __syncthreads();
  }
  int g  = lane >> 4;
  int nl = lane & 15;
  int n = nbase + wv * 16 + nl;
  float bb = (bias != nullptr && n < N) ? bias[n] : 0.f;
#pragma unroll
  for (int r = 0; r < 8; ++r) {
    int m = mtile + 8 * g + r;
    if (n < N) {
      float v = acc0[r] + acc1[r] + bb;
      if (OMODE == 0) Out[(size_t)m * ldo + ocol + n] = v;
      else {
        int b = m >> 14, p = m & (HWp - 1);
        Out[(size_t)(b * N + n) * HWp + p] = v;
      }
    }
  }
}

// ---------------------------------------------------------------------------
// 3x3 conv (pad=1) as implicit GEMM, bf16 WMMA. X,Out NCHW.
// Cin = Cout = 64 (hardcoded), K = 576. Branchless boundary handling.
// ---------------------------------------------------------------------------
__global__ __launch_bounds__(128) void k_conv3(
    const float* __restrict__ X, const float* __restrict__ Wt,
    const float* __restrict__ bias, float* __restrict__ Out,
    int Hd, int Wd, int WT)
{
  __shared__ __align__(32) unsigned lA[512];
  __shared__ __align__(32) unsigned lB[2048];
  int tid  = threadIdx.x;
  int lane = tid & 31;
  int wv   = tid >> 5;
  int bx = blockIdx.x;
  int wt = bx % WT;
  int h  = (bx / WT) % Hd;
  int b  = bx / (WT * Hd);
  const int K = 576;            // 9 * 64
  v8f acc0 = {}, acc1 = {};
  const v16bf* pA = reinterpret_cast<const v16bf*>(lA);
  const v16bf* pB = reinterpret_cast<const v16bf*>(lB);
  for (int kk = 0; kk < K; kk += 64) {
#pragma unroll
    for (int i = 0; i < 4; ++i) {
      int p = tid + i * 128;
      int j = p >> 8, rem = p & 255;
      int l = rem >> 3, r = rem & 7;
      int m = l & 15, g = l >> 4;
      int ka0 = ((r < 4) ? (2 * r) : (2 * r + 8)) + 8 * g;
      int k0 = kk + j * 32 + ka0;          // even; pair shares one tap (Cin=64)
      int tap = k0 >> 6, ci = k0 & 63;
      int t3 = (tap * 11) >> 5;            // tap/3 for tap in [0,8]
      int dy = t3 - 1, dx = tap - 3 * t3 - 1;
      int hp = h + dy, wp = wt * 16 + m + dx;
      bool ok = ((unsigned)hp < (unsigned)Hd) && ((unsigned)wp < (unsigned)Wd);
      int hpc = ok ? hp : 0, wpc = ok ? wp : 0;
      int idx = ((b * 64 + ci) * Hd + hpc) * Wd + wpc;
      float v0 = X[idx];
      float v1 = X[idx + Hd * Wd];
      if (!ok) { v0 = 0.f; v1 = 0.f; }
      lA[j * 256 + l * 8 + r] = packbf(v0, v1);
    }
#pragma unroll
    for (int i = 0; i < 16; ++i) {
      int p = tid + i * 128;
      int j = p >> 10, rem = p & 1023;
      int w = rem >> 8, l = (rem >> 3) & 31, q = rem & 7;
      int kb0 = ((q < 4) ? (2 * q) : (2 * q + 8)) + 8 * (l >> 4);
      int co = w * 16 + (l & 15);
      int k0 = kk + j * 32 + kb0;
      int tap = k0 >> 6, ci = k0 & 63;
      int idx = ((co * 64) + ci) * 9 + tap;
      lB[j * 1024 + (w * 32 + l) * 8 + q] = packbf(Wt[idx], Wt[idx + 9]);
    }
    __syncthreads();
    v16bf av, bv;
    av = pA[lane];            bv = pB[wv * 32 + lane];
    acc0 = __builtin_amdgcn_wmma_f32_16x16x32_bf16(false, av, false, bv,
                                                   (short)0, acc0, false, false);
    av = pA[32 + lane];       bv = pB[128 + wv * 32 + lane];
    acc1 = __builtin_amdgcn_wmma_f32_16x16x32_bf16(false, av, false, bv,
                                                   (short)0, acc1, false, false);
    __syncthreads();
  }
  int g  = lane >> 4;
  int nl = lane & 15;
  int co = wv * 16 + nl;
  float bb = (bias != nullptr) ? bias[co] : 0.f;
#pragma unroll
  for (int r = 0; r < 8; ++r) {
    int wo = wt * 16 + 8 * g + r;
    if (wo < Wd)
      Out[((size_t)(b * 64 + co) * Hd + h) * Wd + wo] = acc0[r] + acc1[r] + bb;
  }
}

// ------------------------------- FFT kernels -------------------------------
__global__ void k_dft_row_fwd(const float* __restrict__ x,
                              float* __restrict__ re, float* __restrict__ im) {
  int h = blockIdx.x, c = blockIdx.y, b = blockIdx.z;
  __shared__ float s[128];
  int t = threadIdx.x;
  const float* row = x + (((size_t)b * Cc + c) * Hh + h) * Ww;
  s[t] = row[t] + 1e-8f;
  __syncthreads();
  if (t < WF) {
    float sr = 0.f, si = 0.f;
    float w = -2.f * PI_F * (float)t / 128.f;
    for (int n = 0; n < 128; ++n) {
      float sv, cv; __sincosf(w * (float)n, &sv, &cv);
      sr += s[n] * cv; si += s[n] * sv;
    }
    size_t o = (((size_t)b * Cc + c) * Hh + h) * WF + t;
    re[o] = sr; im[o] = si;
  }
}

__global__ void k_dft_col_fwd(const float* __restrict__ re, const float* __restrict__ im,
                              float* __restrict__ amp, float* __restrict__ pha) {
  int kx = blockIdx.x, c = blockIdx.y, b = blockIdx.z;
  __shared__ float sr[128], si[128];
  int t = threadIdx.x;
  size_t base = ((size_t)b * Cc + c) * Hh * WF + kx;
  sr[t] = re[base + (size_t)t * WF];
  si[t] = im[base + (size_t)t * WF];
  __syncthreads();
  float ar = 0.f, ai = 0.f;
  float w = -2.f * PI_F * (float)t / 128.f;
  for (int h = 0; h < 128; ++h) {
    float sv, cv; __sincosf(w * (float)h, &sv, &cv);
    ar += sr[h] * cv - si[h] * sv;
    ai += sr[h] * sv + si[h] * cv;
  }
  size_t o = base + (size_t)t * WF;
  amp[o] = sqrtf(ar * ar + ai * ai);
  pha[o] = atan2f(ai, ar);
}

__global__ void k_dft_col_inv(const float* __restrict__ re, const float* __restrict__ im,
                              float* __restrict__ ro, float* __restrict__ io) {
  int kx = blockIdx.x, c = blockIdx.y, b = blockIdx.z;
  __shared__ float sr[128], si[128];
  int t = threadIdx.x;
  size_t base = ((size_t)b * Cc + c) * Hh * WF + kx;
  sr[t] = re[base + (size_t)t * WF];
  si[t] = im[base + (size_t)t * WF];
  __syncthreads();
  float ar = 0.f, ai = 0.f;
  float w = 2.f * PI_F * (float)t / 128.f;
  for (int h = 0; h < 128; ++h) {
    float sv, cv; __sincosf(w * (float)h, &sv, &cv);
    ar += sr[h] * cv - si[h] * sv;
    ai += sr[h] * sv + si[h] * cv;
  }
  size_t o = base + (size_t)t * WF;
  ro[o] = ar * (1.f / 128.f);
  io[o] = ai * (1.f / 128.f);
}

__global__ void k_dft_row_inv_abs(const float* __restrict__ re, const float* __restrict__ im,
                                  float* __restrict__ out) {
  int h = blockIdx.x, c = blockIdx.y, b = blockIdx.z;
  __shared__ float sr[WF], si[WF];
  int t = threadIdx.x;
  size_t base = (((size_t)b * Cc + c) * Hh + h) * WF;
  if (t < WF) { sr[t] = re[base + t]; si[t] = im[base + t]; }
  __syncthreads();
  float acc = 0.f;
  for (int k = 0; k < WF; ++k) {
    float wt = (k == 0 || k == 64) ? 1.f : 2.f;
    float a = 2.f * PI_F * (float)k * (float)t / 128.f;
    float sv, cv; __sincosf(a, &sv, &cv);
    acc += wt * (sr[k] * cv - si[k] * sv);
  }
  out[(((size_t)b * Cc + c) * Hh + h) * Ww + t] = fabsf(acc * (1.f / 128.f));
}

// ---------------------------- elementwise kernels --------------------------
__global__ void k_lrelu(float* p, int n) {
  int i = blockIdx.x * blockDim.x + threadIdx.x;
  if (i < n) { float v = p[i]; p[i] = v > 0.f ? v : 0.1f * v; }
}
__global__ void k_addto(float* __restrict__ dst, const float* __restrict__ a, int n) {
  int i = blockIdx.x * blockDim.x + threadIdx.x;
  if (i < n) dst[i] += a[i];
}
__global__ void k_recon(const float* __restrict__ amp, const float* __restrict__ pha,
                        float* __restrict__ re, float* __restrict__ im, int n) {
  int i = blockIdx.x * blockDim.x + threadIdx.x;
  if (i < n) {
    float a = amp[i], p = pha[i];
    re[i] = a * __cosf(p) + 2e-8f;   // 1e-8 (real) + 1e-8 (scalar add)
    im[i] = a * __sinf(p) + 1e-8f;
  }
}
__global__ void k_nan(float* p, int n) {
  int i = blockIdx.x * blockDim.x + threadIdx.x;
  if (i < n) {
    float v = p[i];
    if (v != v || __builtin_isinf(v)) p[i] = 1e-5f;
  }
}
__global__ void k_bnstats(const float* __restrict__ x, float* __restrict__ mv, int C) {
  int c = blockIdx.x; int t = threadIdx.x;
  __shared__ float s1[256], s2[256];
  float a = 0.f, a2 = 0.f;
  const int total = Bn * HWp;
  for (int j = t; j < total; j += 256) {
    int b = j >> 14, p = j & (HWp - 1);
    float v = x[((size_t)(b * C + c)) * HWp + p];
    a += v; a2 += v * v;
  }
  s1[t] = a; s2[t] = a2; __syncthreads();
  for (int off = 128; off; off >>= 1) {
    if (t < off) { s1[t] += s1[t + off]; s2[t] += s2[t + off]; }
    __syncthreads();
  }
  if (t == 0) {
    float mean = s1[0] / (float)total;
    mv[c] = mean;
    mv[C + c] = s2[0] / (float)total - mean * mean;
  }
}
__global__ void k_bnapply(const float* __restrict__ x, const float* __restrict__ mv,
                          const float* __restrict__ g, const float* __restrict__ be,
                          const float* __restrict__ add, float* __restrict__ out,
                          int relu, int n) {
  int i = blockIdx.x * blockDim.x + threadIdx.x;
  if (i < n) {
    int c = (i >> 14) & 63;     // (i / HWp) % 64
    float v = (x[i] - mv[c]) * rsqrtf(mv[64 + c] + 1e-5f) * g[c] + be[c];
    if (add) v += add[i];
    if (relu) v = v > 0.f ? v : 0.f;
    out[i] = v;
  }
}
__global__ void k_interleave(const float* __restrict__ m1, const float* __restrict__ m2,
                             float* __restrict__ xf, int n) {
  int i = blockIdx.x * blockDim.x + threadIdx.x;
  if (i < n) {
    int ch = i & 127;
    int m  = i >> 7;                 // b*HWp + p
    int b  = m >> 14, p = m & (HWp - 1);
    int cc = ch >> 1;
    const float* src = (ch & 1) ? m2 : m1;
    xf[i] = src[((size_t)(b * Cc + cc)) * HWp + p];
  }
}
__global__ __launch_bounds__(128) void k_layernorm(const float* __restrict__ in,
                                                   const float* __restrict__ g,
                                                   const float* __restrict__ be,
                                                   float* __restrict__ out) {
  int m = blockIdx.x, t = threadIdx.x;
  __shared__ float s1[128], s2[128];
  float v = in[(size_t)m * 128 + t];
  s1[t] = v; s2[t] = v * v; __syncthreads();
  for (int off = 64; off; off >>= 1) {
    if (t < off) { s1[t] += s1[t + off]; s2[t] += s2[t + off]; }
    __syncthreads();
  }
  float mean = s1[0] * (1.f / 128.f);
  float var  = s2[0] * (1.f / 128.f) - mean * mean;
  out[(size_t)m * 128 + t] = (v - mean) * rsqrtf(var + 1e-5f) * g[t] + be[t];
}
__global__ void k_dwconv(const float* __restrict__ xz, const float* __restrict__ cw,
                         const float* __restrict__ cb, float* __restrict__ xt, int n) {
  int i = blockIdx.x * blockDim.x + threadIdx.x;
  if (i < n) {
    int d = i & 63, m = i >> 6;          // m = b*L + l
    int b = m >> 14, l = m & (Ltot - 1);
    float s = cb[d];
#pragma unroll
    for (int j = 0; j < 4; ++j) {
      int l2 = l - 3 + j;
      if (l2 >= 0) s += xz[((size_t)(b * Ltot + l2)) * 128 + d] * cw[d * 4 + j];
    }
    xt[(size_t)m * 64 + d] = siluf(s);
  }
}
// fused softplus(delta) + selective scan; one wave handles two (b,d) recurrences
__global__ __launch_bounds__(32) void k_scan(
    const float* __restrict__ xdbl4, const float* __restrict__ xt4,
    const float* __restrict__ dtw, const float* __restrict__ dtb,
    const float* __restrict__ A_log, float* __restrict__ y4) {
  int bid = blockIdx.x;
  int c  = bid >> 6;          // chunk
  int r  = bid & 63;
  int b  = r >> 5;
  int dp = r & 31;
  int lane = threadIdx.x;
  int n = lane & 15;
  int d = dp * 2 + (lane >> 4);
  const float* xd = xdbl4 + (size_t)c * Mrows * 34;
  const float* xt = xt4   + (size_t)c * Mrows * 64;
  float*       y  = y4    + (size_t)c * Mrows * 64;
  float A  = -__expf(A_log[d * 16 + n]);
  float w0 = dtw[d * 2 + 0], w1 = dtw[d * 2 + 1], tb = dtb[d];
  float h = 0.f;
  for (int l = 0; l < Ltot; ++l) {
    size_t m = (size_t)b * Ltot + l;
    float dt0 = xd[m * 34 + 0], dt1 = xd[m * 34 + 1];
    float dv  = dt0 * w0 + dt1 * w1 + tb;
    float del = dv > 20.f ? dv : log1pf(__expf(dv));
    float bv  = xd[m * 34 + 2 + n];
    float cv  = xd[m * 34 + 18 + n];
    float xl  = xt[m * 64 + d];
    float dA  = __expf(del * A);
    h = dA * h + del * bv * xl;
    float contrib = h * cv;
#pragma unroll
    for (int off = 8; off; off >>= 1) contrib += __shfl_xor(contrib, off, 32);
    if (n == 0) y[m * 64 + d] = contrib;
  }
}
__global__ void k_gate(float* __restrict__ y, const float* __restrict__ xt,
                       const float* __restrict__ Dp, const float* __restrict__ xz, int n) {
  int i = blockIdx.x * blockDim.x + threadIdx.x;
  if (i < n) {
    int d = i & 63, m = i >> 6;
    float v = y[i] + xt[i] * Dp[d];
    float z = xz[(size_t)m * 128 + 64 + d];
    y[i] = v * siluf(z);
  }
}
__global__ void k_skipadd(float* __restrict__ xm, const float* __restrict__ xn,
                          const float* __restrict__ skip, int n) {
  int i = blockIdx.x * blockDim.x + threadIdx.x;
  if (i < n) xm[i] += skip[0] * xn[i];
}

// ---------------------------------------------------------------------------
extern "C" void kernel_launch(void* const* d_in, const int* in_sizes, int n_in,
                              void* d_out, int out_size, void* d_ws, size_t ws_size,
                              hipStream_t stream) {
  (void)in_sizes; (void)n_in; (void)out_size; (void)ws_size;
  const float* x       = (const float*)d_in[0];
  const float* amp_w1  = (const float*)d_in[1];
  const float* amp_b1  = (const float*)d_in[2];
  const float* amp_w2  = (const float*)d_in[3];
  const float* amp_b2  = (const float*)d_in[4];
  const float* pha_w1  = (const float*)d_in[5];
  const float* pha_b1  = (const float*)d_in[6];
  const float* pha_w2  = (const float*)d_in[7];
  const float* pha_b2  = (const float*)d_in[8];
  const float* adj_w   = (const float*)d_in[9];
  const float* adj_b   = (const float*)d_in[10];
  const float* e_w1    = (const float*)d_in[11];
  const float* e_b1    = (const float*)d_in[12];
  const float* e_g1    = (const float*)d_in[13];
  const float* e_be1   = (const float*)d_in[14];
  const float* e_w2    = (const float*)d_in[15];
  const float* e_b2    = (const float*)d_in[16];
  const float* e_g2    = (const float*)d_in[17];
  const float* e_be2   = (const float*)d_in[18];
  const float* e_w3    = (const float*)d_in[19];
  const float* e_b3    = (const float*)d_in[20];
  const float* ln_g    = (const float*)d_in[21];
  const float* ln_b    = (const float*)d_in[22];
  const float* in_proj = (const float*)d_in[23];
  const float* conv_w  = (const float*)d_in[24];
  const float* conv_b  = (const float*)d_in[25];
  const float* x_proj  = (const float*)d_in[26];
  const float* dt_w    = (const float*)d_in[27];
  const float* dt_b    = (const float*)d_in[28];
  const float* A_log   = (const float*)d_in[29];
  const float* Dp      = (const float*)d_in[30];
  const float* out_prj = (const float*)d_in[31];
  const float* prj_w   = (const float*)d_in[32];
  const float* prj_b   = (const float*)d_in[33];
  const float* skip    = (const float*)d_in[34];
  float* out = (float*)d_out;
  float* ws  = (float*)d_ws;

  // workspace layout (floats); scratch region S is phase-aliased
  const size_t M1O = 0;
  const size_t M2O = M1O + NSp;
  const size_t XN  = M2O + NSp;
  const size_t XM  = XN + (size_t)Mrows * 128;
  const size_t S   = XM + (size_t)Mrows * 128;
  // phase A/B aliases inside S
  const size_t RE = S, IM = RE + NFr, AMP = IM + NFr, PHA = AMP + NFr;
  const size_t T1 = PHA + NFr, T2 = T1 + NFr;
  const size_t M1S = T2 + NFr;
  const size_t E1 = M1S + NSp, E2 = E1 + NSp, E3 = E2 + NSp;
  // phase C alias
  const size_t XF = S;
  // phase D aliases (freq/enc buffers are dead)
  const size_t XZ4 = S;
  const size_t XT4 = XZ4 + (size_t)4 * Mrows * 128;
  const size_t XD4 = XT4 + (size_t)4 * Mrows * 64;
  const size_t YB4 = XD4 + (size_t)4 * Mrows * 34;
  // phase E alias
  const size_t XN2 = S;
  const size_t STATS = YB4 + (size_t)4 * Mrows * 64;

  dim3 blk256(256);
  auto G = [](int n) { return dim3((unsigned)((n + 255) / 256)); };

  // ---------------- Phase A: freq_filter -> M1O ----------------
  k_dft_row_fwd<<<dim3(Hh, Cc, Bn), 128, 0, stream>>>(x, ws + RE, ws + IM);
  k_dft_col_fwd<<<dim3(WF, Cc, Bn), 128, 0, stream>>>(ws + RE, ws + IM, ws + AMP, ws + PHA);
  // amp mask net (128 x 65 images)
  k_conv3<<<dim3(Bn * Hh * 5, 1), 128, 0, stream>>>(ws + AMP, amp_w1, amp_b1, ws + T1, Hh, WF, 5);
  k_lrelu<<<G(NFr), blk256, 0, stream>>>(ws + T1, NFr);
  k_conv3<<<dim3(Bn * Hh * 5, 1), 128, 0, stream>>>(ws + T1, amp_w2, amp_b2, ws + T2, Hh, WF, 5);
  k_addto<<<G(NFr), blk256, 0, stream>>>(ws + AMP, ws + T2, NFr);
  // pha mask net
  k_conv3<<<dim3(Bn * Hh * 5, 1), 128, 0, stream>>>(ws + PHA, pha_w1, pha_b1, ws + T1, Hh, WF, 5);
  k_lrelu<<<G(NFr), blk256, 0, stream>>>(ws + T1, NFr);
  k_conv3<<<dim3(Bn * Hh * 5, 1), 128, 0, stream>>>(ws + T1, pha_w2, pha_b2, ws + T2, Hh, WF, 5);
  k_addto<<<G(NFr), blk256, 0, stream>>>(ws + PHA, ws + T2, NFr);
  // reconstruct + inverse FFT + abs
  k_recon<<<G(NFr), blk256, 0, stream>>>(ws + AMP, ws + PHA, ws + RE, ws + IM, NFr);
  k_dft_col_inv<<<dim3(WF, Cc, Bn), 128, 0, stream>>>(ws + RE, ws + IM, ws + T1, ws + T2);
  k_dft_row_inv_abs<<<dim3(Hh, Cc, Bn), 128, 0, stream>>>(ws + T1, ws + T2, ws + M1S);
  k_addto<<<G(NSp), blk256, 0, stream>>>(ws + M1S, x, NSp);
  // adj 1x1 conv (implicit GEMM, NCHW in/out)
  k_gemm<1, 1><<<dim3(Mrows / 16, 1), 128, 0, stream>>>(ws + M1S, adj_w, adj_b, ws + M1O,
      Mrows, Cc, Cc, 0, 0, 0, 0);
  k_nan<<<G(NSp), blk256, 0, stream>>>(ws + M1O, NSp);

  // ---------------- Phase B: double_conv -> M2O ----------------
  k_conv3<<<dim3(Bn * Hh * 8, 1), 128, 0, stream>>>(x, e_w1, e_b1, ws + E1, Hh, Ww, 8);
  k_bnstats<<<dim3(Cc), blk256, 0, stream>>>(ws + E1, ws + STATS, Cc);
  k_bnapply<<<G(NSp), blk256, 0, stream>>>(ws + E1, ws + STATS, e_g1, e_be1, nullptr, ws + E1, 1, NSp);
  k_conv3<<<dim3(Bn * Hh * 8, 1), 128, 0, stream>>>(ws + E1, e_w2, e_b2, ws + E2, Hh, Ww, 8);
  k_bnstats<<<dim3(Cc), blk256, 0, stream>>>(ws + E2, ws + STATS + 128, Cc);
  k_gemm<1, 1><<<dim3(Mrows / 16, 1), 128, 0, stream>>>(x, e_w3, e_b3, ws + E3,
      Mrows, Cc, Cc, 0, 0, 0, 0);
  k_bnapply<<<G(NSp), blk256, 0, stream>>>(ws + E2, ws + STATS + 128, e_g2, e_be2, ws + E3, ws + M2O, 1, NSp);

  // ---------------- Phase C: interleave + layernorm ----------------
  k_interleave<<<G(Mrows * 128), blk256, 0, stream>>>(ws + M1O, ws + M2O, ws + XF, Mrows * 128);
  k_layernorm<<<dim3(Mrows), 128, 0, stream>>>(ws + XF, ln_g, ln_b, ws + XN);

  // ---------------- Phase D: 4 mamba chunks ----------------
  for (int c = 0; c < 4; ++c) {
    float* xz = ws + XZ4 + (size_t)c * Mrows * 128;
    float* xt = ws + XT4 + (size_t)c * Mrows * 64;
    float* xd = ws + XD4 + (size_t)c * Mrows * 34;
    // in_proj: (B*L,32) @ (128,32)^T -> xz (B*L,128)
    k_gemm<0, 0><<<dim3(Mrows / 16, 2), 128, 0, stream>>>(ws + XN, in_proj, nullptr, xz,
        Mrows, 128, 32, 128, 32 * c, 128, 0);
    k_dwconv<<<G(Mrows * 64), blk256, 0, stream>>>(xz, conv_w, conv_b, xt, Mrows * 64);
    // x_proj: (B*L,64) @ (34,64)^T -> xdbl (B*L,34)
    k_gemm<0, 0><<<dim3(Mrows / 16, 1), 128, 0, stream>>>(xt, x_proj, nullptr, xd,
        Mrows, 34, 64, 64, 0, 34, 0);
  }
  // fused softplus + selective scan over all 4 chunks (256 waves)
  k_scan<<<dim3(256), 32, 0, stream>>>(ws + XD4, ws + XT4, dt_w, dt_b, A_log, ws + YB4);
  for (int c = 0; c < 4; ++c) {
    float* xz = ws + XZ4 + (size_t)c * Mrows * 128;
    float* xt = ws + XT4 + (size_t)c * Mrows * 64;
    float* yb = ws + YB4 + (size_t)c * Mrows * 64;
    k_gate<<<G(Mrows * 64), blk256, 0, stream>>>(yb, xt, Dp, xz, Mrows * 64);
    // out_proj: (B*L,64) @ (32,64)^T -> XM columns [32c..32c+32)
    k_gemm<0, 0><<<dim3(Mrows / 16, 1), 128, 0, stream>>>(yb, out_prj, nullptr, ws + XM,
        Mrows, 32, 64, 64, 0, 128, 32 * c);
  }
  k_skipadd<<<G(Mrows * 128), blk256, 0, stream>>>(ws + XM, ws + XN, skip, Mrows * 128);

  // ---------------- Phase E: layernorm + final projection ----------------
  k_layernorm<<<dim3(Mrows), 128, 0, stream>>>(ws + XM, ln_g, ln_b, ws + XN2);
  k_gemm<0, 1><<<dim3(Mrows / 16, 1), 128, 0, stream>>>(ws + XN2, prj_w, prj_b, out,
      Mrows, 64, 128, 128, 0, 0, 0);
}
